// PointerGeneratorModel_61641370632430
// MI455X (gfx1250) — matrix operand order repaired and checked by reading, not compiled, not run on hardware
//
#include <hip/hip_runtime.h>
#include <hip/hip_bf16.h>
#include <hip/hip_fp16.h>

// Pointer-generator decoder for MI455X (gfx1250, wave32, WMMA).
// Heavy GEMMs (enc_proj, vocab projection) use v_wmma_f32_16x16x32_f16 with
// f32 accumulate; weights converted to f16 once per call (halves L2 traffic,
// W_out2 stays L2-resident at 25.6MB). Sequential T=64 decode is a per-step
// 5-kernel chain on `stream` (graph-capturable, deterministic).

#define Bb    8
#define Ss    1024
#define Tt    64
#define Ee    128
#define Hh    256
#define Vv    50000
#define OOVv  500
#define VEXT  (Vv + OOVv)
#define TWOH  512
#define FOURH 1024
#define XDIM  (Ee + TWOH)   // 640  (emb ++ ctx)
#define KHC   (Hh + TWOH)   // 768  (h ++ ctx)

typedef __attribute__((ext_vector_type(16))) _Float16 v16h;
typedef __attribute__((ext_vector_type(8)))  _Float16 v8h;
typedef __attribute__((ext_vector_type(8)))  float    v8f;

// ---------------- WMMA fragment loaders (row-major f16, K contiguous) -------
// A 16x32 f16: lane m (0..15): K = hi*16 .. hi*16+15, hi=lane>>4
__device__ __forceinline__ v16h load_a_frag(const _Float16* base, int ld, int kc) {
    int lane = threadIdx.x & 31;
    int m    = lane & 15;
    int hi   = lane >> 4;
    const _Float16* p = base + (size_t)m * ld + kc + hi * 16;
    return *(const v16h*)p;
}
// B 32x16 f16: lane n (0..15) holds K=0..15; lanes 16..31 hold K=16..31
__device__ __forceinline__ v16h load_b_frag(const _Float16* base, int ld, int kc) {
    int lane = threadIdx.x & 31;
    int n    = lane & 15;
    int hi   = lane >> 4;
    const _Float16* p = base + (size_t)n * ld + kc + hi * 16;
    return *(const v16h*)p;
}

// ---------------- small utility kernels ------------------------------------
__global__ void k_f32_to_f16(const float* __restrict__ src,
                             _Float16* __restrict__ dst, int n) {
    int i = blockIdx.x * blockDim.x + threadIdx.x;
    if (i < n) dst[i] = (_Float16)src[i];
}

__global__ void k_zero(float* __restrict__ p, int n) {
    int i = blockIdx.x * blockDim.x + threadIdx.x;
    if (i < n) p[i] = 0.0f;
}

// ---------------- enc_proj = einsum('bse,he->bsh') via WMMA -----------------
// M = B*S = 8192 rows of enc_f16 (K=512), B-matrix = W_h rows (K contiguous).
// 512 M-tiles x 16 N-tiles = 8192 tiles; 8 waves/block -> 1024 blocks.
__global__ __launch_bounds__(256) void k_encproj(
    const _Float16* __restrict__ encf, const _Float16* __restrict__ whf,
    float* __restrict__ encproj) {
    int wave = threadIdx.x >> 5;
    int tile = blockIdx.x * 8 + wave;
    int mt = tile >> 4, nt = tile & 15;
    const _Float16* Ab = encf + (size_t)(mt * 16) * TWOH;
    const _Float16* Bbm = whf + (size_t)(nt * 16) * TWOH;
    v8f acc = {};
#pragma unroll 4
    for (int kc = 0; kc < TWOH; kc += 32) {
        v16h a = load_a_frag(Ab, TWOH, kc);
        v16h b = load_b_frag(Bbm, TWOH, kc);
        acc = __builtin_amdgcn_wmma_f32_16x16x32_f16(false, a, false, b,
                                                     (short)0, acc, false, false);
    }
    int lane = threadIdx.x & 31;
    int n = lane & 15, hi = lane >> 4;
    int m0 = mt * 16, n0 = nt * 16;
#pragma unroll
    for (int r = 0; r < 8; ++r)
        encproj[(size_t)(m0 + r + 8 * hi) * Hh + n0 + n] = acc[r];
}

// ---------------- per-step: LSTM cell + dh = h @ W_d^T ----------------------
__global__ __launch_bounds__(1024) void k_lstm(
    const float* __restrict__ emb, const int* __restrict__ dec_ids,
    const float* __restrict__ W_ih, const float* __restrict__ W_hh,
    const float* __restrict__ b_lstm, const float* __restrict__ W_d,
    float* __restrict__ h_st, float* __restrict__ c_st,
    const float* __restrict__ ctx_in,
    float* __restrict__ dh, float* __restrict__ xsave, int t) {
    __shared__ float xs[Bb][XDIM];   // 20 KB
    __shared__ float hs[Bb][Hh];     //  8 KB
    __shared__ float gb[Bb][FOURH];  // 32 KB
    int tid = threadIdx.x;
    for (int i = tid; i < Bb * XDIM; i += 1024) {
        int b = i / XDIM, k = i % XDIM;
        float val;
        if (k < Ee) {
            int id = dec_ids[b * Tt + t];
            val = emb[(size_t)id * Ee + k];
        } else {
            val = ctx_in[b * TWOH + (k - Ee)];
        }
        xs[b][k] = val;
        xsave[i] = val;  // saved for p_gen ([... , x])
    }
    for (int i = tid; i < Bb * Hh; i += 1024) hs[i / Hh][i % Hh] = h_st[i];
    __syncthreads();
    {   // gates[b][j] for j = tid, all 8 batches
        int j = tid;
        float acc[Bb];
        float bl = b_lstm[j];
#pragma unroll
        for (int b = 0; b < Bb; ++b) acc[b] = bl;
        const float* wih = W_ih + (size_t)j * XDIM;
        for (int k = 0; k < XDIM; ++k) {
            float w = wih[k];
#pragma unroll
            for (int b = 0; b < Bb; ++b) acc[b] += w * xs[b][k];
        }
        const float* whh = W_hh + (size_t)j * Hh;
        for (int k = 0; k < Hh; ++k) {
            float w = whh[k];
#pragma unroll
            for (int b = 0; b < Bb; ++b) acc[b] += w * hs[b][k];
        }
#pragma unroll
        for (int b = 0; b < Bb; ++b) gb[b][j] = acc[b];
    }
    __syncthreads();
    for (int i = tid; i < Bb * Hh; i += 1024) {  // cell update (torch gate order)
        int b = i / Hh, j = i % Hh;
        float ig = gb[b][j], fg = gb[b][Hh + j];
        float gg = gb[b][2 * Hh + j], og = gb[b][3 * Hh + j];
        float si = 1.0f / (1.0f + expf(-ig));
        float sf = 1.0f / (1.0f + expf(-fg));
        float so = 1.0f / (1.0f + expf(-og));
        float c = sf * c_st[i] + si * tanhf(gg);
        float h = so * tanhf(c);
        c_st[i] = c; h_st[i] = h;
        hs[b][j] = h;  // reuse for W_d
    }
    __syncthreads();
    if (tid < Hh) {  // dh[b][tid] = sum_k W_d[tid][k] * h[b][k]
        const float* wd = W_d + (size_t)tid * Hh;
        float acc[Bb];
#pragma unroll
        for (int b = 0; b < Bb; ++b) acc[b] = 0.0f;
        for (int k = 0; k < Hh; ++k) {
            float w = wd[k];
#pragma unroll
            for (int b = 0; b < Bb; ++b) acc[b] += w * hs[b][k];
        }
#pragma unroll
        for (int b = 0; b < Bb; ++b) dh[b * Hh + tid] = acc[b];
    }
}

// ---------------- per-step (fused): coverage energies -> softmax -> ctx -----
// One block per batch row b (8 blocks x 512 threads). Scores never touch
// global memory: each thread computes the tanh-energy dot for two positions,
// block-reduces max/sum in LDS, updates coverage, and accumulates the
// context vector with fully coalesced reads of encoder_outputs.
__global__ __launch_bounds__(512) void k_attn(
    const float* __restrict__ encproj, const float* __restrict__ dh,
    const float* __restrict__ w_c, const float* __restrict__ vvec,
    const int* __restrict__ mask, const float* __restrict__ enc,
    float* __restrict__ attn, float* __restrict__ cov, float* __restrict__ ctx) {
    __shared__ float dhs[Hh], wcs[Hh], vs[Hh];
    __shared__ float al[Ss];
    __shared__ float red[512];
    int b = blockIdx.x, tid = threadIdx.x;
    for (int j = tid; j < Hh; j += 512) {
        dhs[j] = dh[b * Hh + j]; wcs[j] = w_c[j]; vs[j] = vvec[j];
    }
    __syncthreads();
    // energies + scores for positions s0 = tid, s1 = tid + 512
    int i0 = b * Ss + tid, i1 = i0 + 512;
    float cv0 = cov[i0], cv1 = cov[i1];
    const float* ep0 = encproj + (size_t)i0 * Hh;
    const float* ep1 = encproj + (size_t)i1 * Hh;
    float s0 = 0.0f, s1 = 0.0f;
    for (int j = 0; j < Hh; ++j) {
        float vj = vs[j], dj = dhs[j], wj = wcs[j];
        s0 += vj * tanhf(ep0[j] + dj + cv0 * wj);
        s1 += vj * tanhf(ep1[j] + dj + cv1 * wj);
    }
    if (mask[i0] == 0) s0 -= 10000.0f;
    if (mask[i1] == 0) s1 -= 10000.0f;
    // softmax over S=1024
    red[tid] = fmaxf(s0, s1);
    __syncthreads();
    for (int off = 256; off > 0; off >>= 1) {
        if (tid < off) red[tid] = fmaxf(red[tid], red[tid + off]);
        __syncthreads();
    }
    float mx = red[0];
    __syncthreads();
    float e0 = expf(s0 - mx), e1 = expf(s1 - mx);
    red[tid] = e0 + e1;
    __syncthreads();
    for (int off = 256; off > 0; off >>= 1) {
        if (tid < off) red[tid] += red[tid + off];
        __syncthreads();
    }
    float inv = 1.0f / red[0];
    e0 *= inv; e1 *= inv;
    al[tid] = e0; al[tid + 512] = e1;
    attn[i0] = e0; attn[i1] = e1;
    cov[i0] = cv0 + e0; cov[i1] = cv1 + e1;
    __syncthreads();
    // ctx[b][d] = sum_s attn[b][s] * enc[b][s][d], d = tid (0..511), coalesced
    float acc = 0.0f;
    const float* eb = enc + (size_t)b * Ss * TWOH + tid;
    for (int s = 0; s < Ss; ++s) acc += al[s] * eb[(size_t)s * TWOH];
    ctx[b * TWOH + tid] = acc;
}

// ---------------- per-step: hidden = [h,ctx]@W_out1^T + b, p_gen ------------
__global__ __launch_bounds__(256) void k_outproj_small(
    const float* __restrict__ h, const float* __restrict__ ctx,
    const float* __restrict__ xsave,
    const float* __restrict__ W_out1, const float* __restrict__ b_out1,
    const float* __restrict__ w_pg, const float* __restrict__ b_pg,
    _Float16* __restrict__ hid16, float* __restrict__ pgen) {
    __shared__ float hc[Bb][KHC];  // 24 KB: [h | ctx_new]
    int tid = threadIdx.x;
    for (int i = tid; i < Bb * KHC; i += 256) {
        int b = i / KHC, k = i % KHC;
        hc[b][k] = (k < Hh) ? h[b * Hh + k] : ctx[b * TWOH + (k - Hh)];
    }
    __syncthreads();
    {   // hidden[b][tid]
        const float* w = W_out1 + (size_t)tid * KHC;
        float acc[Bb];
        float bias = b_out1[tid];
#pragma unroll
        for (int b = 0; b < Bb; ++b) acc[b] = bias;
        for (int k = 0; k < KHC; ++k) {
            float wv = w[k];
#pragma unroll
            for (int b = 0; b < Bb; ++b) acc[b] += wv * hc[b][k];
        }
#pragma unroll
        for (int b = 0; b < Bb; ++b) hid16[b * Hh + tid] = (_Float16)acc[b];
#pragma unroll
        for (int b = Bb; b < 16; ++b) hid16[b * Hh + tid] = (_Float16)0.0f;  // pad M to 16
    }
    if (tid < Bb) {  // p_gen = sigmoid(w_pg . [ctx_new, h, x] + b_pg)
        int b = tid;
        float acc = b_pg[0];
        for (int k = 0; k < TWOH; ++k) acc += w_pg[k] * hc[b][Hh + k];
        for (int k = 0; k < Hh; ++k)   acc += w_pg[TWOH + k] * hc[b][k];
        const float* xb = xsave + b * XDIM;
        for (int k = 0; k < XDIM; ++k) acc += w_pg[TWOH + Hh + k] * xb[k];
        pgen[b] = 1.0f / (1.0f + expf(-acc));
    }
}

// ---------------- per-step: vocab GEMM via WMMA (M=8 pad 16, N=50000, K=256)
__global__ __launch_bounds__(256) void k_vocab(
    const _Float16* __restrict__ hid16, const _Float16* __restrict__ w2f,
    const float* __restrict__ b_out2, float* __restrict__ logits) {
    int wave = threadIdx.x >> 5;
    int nt = blockIdx.x * 8 + wave;           // N-tile index, 3125 tiles total
    if (nt * 16 >= Vv) return;                // whole wave exits -> EXEC stays all-1
    int n0 = nt * 16;
    const _Float16* Bbm = w2f + (size_t)n0 * Hh;  // W_out2 rows = B columns (K contiguous)
    v8f acc = {};
#pragma unroll
    for (int kc = 0; kc < Hh; kc += 32) {
        v16h a = load_a_frag(hid16, Hh, kc);
        v16h b = load_b_frag(Bbm, Hh, kc);
        acc = __builtin_amdgcn_wmma_f32_16x16x32_f16(false, a, false, b,
                                                     (short)0, acc, false, false);
    }
    int lane = threadIdx.x & 31;
    int n = lane & 15, hi = lane >> 4;
    if (hi == 0) {  // rows 0..7 = valid batches (C/D: M = r + 8*hi)
        float bias = b_out2[n0 + n];
#pragma unroll
        for (int r = 0; r < 8; ++r)
            logits[(size_t)r * Vv + n0 + n] = acc[r] + bias;
    }
}

// ---------------- per-step: 50k softmax, mix with copy dist, scatter --------
// exp values are cached in-place into the logits scratch during the sum pass
// (each element is read/written by the same thread in both passes), halving
// the transcendental count on the largest elementwise loop.
__global__ __launch_bounds__(1024) void k_final(
    float* __restrict__ logits, const float* __restrict__ pgen,
    const float* __restrict__ attn, const int* __restrict__ src_ids,
    float* __restrict__ out, int t) {
    __shared__ float red[1024];
    int b = blockIdx.x, tid = threadIdx.x;
    float* lg = logits + (size_t)b * Vv;
    float mx = -3.4e38f;
    for (int j = tid; j < Vv; j += 1024) mx = fmaxf(mx, lg[j]);
    red[tid] = mx;
    __syncthreads();
    for (int off = 512; off > 0; off >>= 1) {
        if (tid < off) red[tid] = fmaxf(red[tid], red[tid + off]);
        __syncthreads();
    }
    mx = red[0];
    __syncthreads();
    float sm = 0.0f;
    for (int j = tid; j < Vv; j += 1024) {
        float e = expf(lg[j] - mx);
        lg[j] = e;              // cache exp in place (same thread re-reads below)
        sm += e;
    }
    red[tid] = sm;
    __syncthreads();
    for (int off = 512; off > 0; off >>= 1) {
        if (tid < off) red[tid] += red[tid + off];
        __syncthreads();
    }
    float scale = pgen[b] / red[0];
    float* ob = out + ((size_t)b * Tt + t) * VEXT;
    for (int j = tid; j < Vv; j += 1024) ob[j] = scale * lg[j];
    for (int j = Vv + tid; j < VEXT; j += 1024) ob[j] = 0.0f;
    __syncthreads();
    if (tid == 0) {  // deterministic sequential scatter-add of copy distribution
        float om = 1.0f - pgen[b];
        for (int s = 0; s < Ss; ++s)
            ob[src_ids[b * Ss + s]] += om * attn[b * Ss + s];
    }
}

// ---------------- host ------------------------------------------------------
extern "C" void kernel_launch(void* const* d_in, const int* in_sizes, int n_in,
                              void* d_out, int out_size, void* d_ws, size_t ws_size,
                              hipStream_t stream) {
    const float* enc     = (const float*)d_in[0];
    const float* emb     = (const float*)d_in[1];
    const float* W_h     = (const float*)d_in[2];
    const float* W_d     = (const float*)d_in[3];
    const float* w_c     = (const float*)d_in[4];
    const float* vvec    = (const float*)d_in[5];
    const float* W_ih    = (const float*)d_in[6];
    const float* W_hh    = (const float*)d_in[7];
    const float* b_lstm  = (const float*)d_in[8];
    const float* W_out1  = (const float*)d_in[9];
    const float* b_out1  = (const float*)d_in[10];
    const float* W_out2  = (const float*)d_in[11];
    const float* b_out2  = (const float*)d_in[12];
    const float* w_pg    = (const float*)d_in[13];
    const float* b_pg    = (const float*)d_in[14];
    const int*   mask    = (const int*)d_in[15];
    const int*   src_ids = (const int*)d_in[16];
    const int*   dec_ids = (const int*)d_in[17];
    float* out = (float*)d_out;

    char* w = (char*)d_ws;
    auto alloc = [&](size_t bytes) -> char* {
        char* p = w;
        w += (bytes + 255) & ~(size_t)255;
        return p;
    };
    _Float16* encf16  = (_Float16*)alloc((size_t)Bb * Ss * TWOH * 2);  // 8 MB
    _Float16* whf16   = (_Float16*)alloc((size_t)Hh * TWOH * 2);
    _Float16* w2f16   = (_Float16*)alloc((size_t)Vv * Hh * 2);         // 25.6 MB
    float* encproj    = (float*)alloc((size_t)Bb * Ss * Hh * 4);       // 8 MB
    float* h_st       = (float*)alloc(Bb * Hh * 4);
    float* c_st       = (float*)alloc(Bb * Hh * 4);
    float* ctx        = (float*)alloc(Bb * TWOH * 4);
    float* cov        = (float*)alloc(Bb * Ss * 4);
    float* dh         = (float*)alloc(Bb * Hh * 4);
    float* xsave      = (float*)alloc(Bb * XDIM * 4);
    float* attn       = (float*)alloc(Bb * Ss * 4);
    _Float16* hid16   = (_Float16*)alloc(16 * Hh * 2);
    float* pgen       = (float*)alloc(256);
    float* logits     = (float*)alloc((size_t)Bb * Vv * 4);            // 1.6 MB

    // one-time (per call) conversions + encoder-side projection
    {
        int n = Bb * Ss * TWOH;
        k_f32_to_f16<<<(n + 255) / 256, 256, 0, stream>>>(enc, encf16, n);
        n = Hh * TWOH;
        k_f32_to_f16<<<(n + 255) / 256, 256, 0, stream>>>(W_h, whf16, n);
        n = Vv * Hh;
        k_f32_to_f16<<<(n + 255) / 256, 256, 0, stream>>>(W_out2, w2f16, n);
    }
    k_encproj<<<(512 * 16) / 8, 256, 0, stream>>>(encf16, whf16, encproj);

    // zero-init recurrent state (h, c, ctx, cov)
    k_zero<<<(Bb * Hh + 255) / 256, 256, 0, stream>>>(h_st, Bb * Hh);
    k_zero<<<(Bb * Hh + 255) / 256, 256, 0, stream>>>(c_st, Bb * Hh);
    k_zero<<<(Bb * TWOH + 255) / 256, 256, 0, stream>>>(ctx, Bb * TWOH);
    k_zero<<<(Bb * Ss + 255) / 256, 256, 0, stream>>>(cov, Bb * Ss);

    for (int t = 0; t < Tt; ++t) {
        k_lstm<<<1, 1024, 0, stream>>>(emb, dec_ids, W_ih, W_hh, b_lstm, W_d,
                                       h_st, c_st, ctx, dh, xsave, t);
        k_attn<<<Bb, 512, 0, stream>>>(encproj, dh, w_c, vvec, mask, enc,
                                       attn, cov, ctx);
        k_outproj_small<<<1, 256, 0, stream>>>(h_st, ctx, xsave, W_out1, b_out1,
                                               w_pg, b_pg, hid16, pgen);
        k_vocab<<<(3125 + 7) / 8, 256, 0, stream>>>(hid16, w2f16, b_out2, logits);
        k_final<<<Bb, 1024, 0, stream>>>(logits, pgen, attn, src_ids, out, t);
    }
    (void)in_sizes; (void)n_in; (void)out_size; (void)ws_size;
}
